// SRL_KT_43387759624761
// MI455X (gfx1250) — compile-verified, matrix operand order checked by reading, not run on hardware
//
#include <hip/hip_runtime.h>

// ---------------------------------------------------------------------------
// SRL knowledge-tracing transformer for MI455X (gfx1250, wave32, WMMA).
// B=8, L=512, E=512, H=8, d=64, NL=2.  All GEMMs: bf16 A/B fragments,
// f32 accumulate via v_wmma_f32_16x16x32_bf16.  Attention is fused
// (scores+softmax+blend+PV) with LDS-resident tiles.  WMMA loops are
// register double-buffered so fragment loads overlap matrix ops; the
// blend-base rows are prefetched (global_prefetch_b8).
// ---------------------------------------------------------------------------

#define B_ 8
#define L_ 512
#define E_ 512
#define H_ 8
#define D_ 64
#define NL_ 2
#define M_ (B_ * L_) /* 4096 token rows */

typedef __attribute__((ext_vector_type(16))) __bf16 bfrag;
typedef __attribute__((ext_vector_type(8))) float v8f;

__device__ __forceinline__ v8f wmma_bf16(bfrag a, bfrag b, v8f c) {
  // D = A(16x32) * B(32x16) + C(16x16 f32)
  return __builtin_amdgcn_wmma_f32_16x16x32_bf16(
      /*neg_a=*/false, a, /*neg_b=*/false, b,
      /*c_mod=*/(short)0, c, /*reuse_a=*/false, /*reuse_b=*/false);
}

__device__ __forceinline__ __bf16 to_bf16(float f) {
  unsigned u = __builtin_bit_cast(unsigned, f);
  unsigned short hs = (unsigned short)((u + 0x7FFFu + ((u >> 16) & 1u)) >> 16);
  return __builtin_bit_cast(__bf16, hs);
}

// 16-bit WMMA fragment layout (ISA 7.12.2, 16-bit A 16x32):
//   lane = m' + 16*half ; slot i holds K = base(pair) + half*8 + (i&1)
__device__ __forceinline__ int frag_k(int half, int slot) {
  int pairIdx = slot >> 1;
  int base = (pairIdx < 4) ? (pairIdx * 2) : ((pairIdx - 4) * 2 + 16);
  return base + half * 8 + (slot & 1);
}

// ------------------------- operand packing kernels -------------------------
// Pack f32 row-major A (MxK) into bf16 fragments: [mt][kt+ktOff][lane][slot]
__global__ __launch_bounds__(256) void pack_a_kernel(
    const float* __restrict__ src, __bf16* __restrict__ dst, int ld, int numMt,
    int numKt, int totalKt, int ktOff, long srcYStride, long dstYStride,
    long srcZStride, long dstZStride) {
  src += (long)blockIdx.y * srcYStride + (long)blockIdx.z * srcZStride;
  dst += (long)blockIdx.y * dstYStride + (long)blockIdx.z * dstZStride;
  int tile = blockIdx.x * 8 + (threadIdx.x >> 5);
  if (tile >= numMt * numKt) return;
  int lane = threadIdx.x & 31;
  int mt = tile / numKt, kt = tile % numKt;
  int mp = lane & 15, half = lane >> 4;
  const float* s = src + (long)(mt * 16 + mp) * ld + kt * 32;
  __bf16* d = dst + ((long)(mt * totalKt + kt + ktOff) * 32 + lane) * 16;
#pragma unroll
  for (int slot = 0; slot < 16; ++slot) d[slot] = to_bf16(s[frag_k(half, slot)]);
}

// Pack B (KxN) into bf16 fragments [nt][kt][lane][slot].
// transpose=1: src is (N x K) row-major (used to form k^T operands).
__global__ __launch_bounds__(256) void pack_b_kernel(
    const float* __restrict__ src, __bf16* __restrict__ dst, int ld, int numNt,
    int numKt, int transpose, long srcYStride, long dstYStride, long srcZStride,
    long dstZStride) {
  src += (long)blockIdx.y * srcYStride + (long)blockIdx.z * srcZStride;
  dst += (long)blockIdx.y * dstYStride + (long)blockIdx.z * dstZStride;
  int tile = blockIdx.x * 8 + (threadIdx.x >> 5);
  if (tile >= numNt * numKt) return;
  int lane = threadIdx.x & 31;
  int nt = tile / numKt, kt = tile % numKt;
  int np = lane & 15, half = lane >> 4;
  __bf16* d = dst + ((long)tile * 32 + lane) * 16;
#pragma unroll
  for (int slot = 0; slot < 16; ++slot) {
    int kk = kt * 32 + frag_k(half, slot);
    int nn = nt * 16 + np;
    float v = transpose ? src[(long)nn * ld + kk] : src[(long)kk * ld + nn];
    d[slot] = to_bf16(v);
  }
}

// ------------------------------ GEMM kernel --------------------------------
// C(f32, MxN, ldc) = A_packed * B_packed (+bias) (+relu). 4 waves / block,
// each wave: one 16-row strip x 64 cols; K-loop is register double-buffered
// so the next iteration's fragment loads overlap the current WMMAs.
__global__ __launch_bounds__(128) void gemm_wmma_kernel(
    const __bf16* __restrict__ A, const __bf16* __restrict__ Bm,
    const float* __restrict__ bias, float* __restrict__ C, int numKt, int ldc,
    int relu) {
  int w = threadIdx.x >> 5, lane = threadIdx.x & 31;
  int half = lane >> 4, nlo = lane & 15;
  int mt = blockIdx.y * 4 + w;
  int nt0 = blockIdx.x * 4;
  v8f acc[4] = {{}, {}, {}, {}};
  const __bf16* aBase = A + ((long)mt * numKt) * 512 + (long)lane * 16;
  const __bf16* bBase = Bm + (long)lane * 16;

  bfrag a_c = *(const bfrag*)(aBase);
  bfrag b_c0 = *(const bfrag*)(bBase + ((long)(nt0 + 0) * numKt) * 512);
  bfrag b_c1 = *(const bfrag*)(bBase + ((long)(nt0 + 1) * numKt) * 512);
  bfrag b_c2 = *(const bfrag*)(bBase + ((long)(nt0 + 2) * numKt) * 512);
  bfrag b_c3 = *(const bfrag*)(bBase + ((long)(nt0 + 3) * numKt) * 512);
  for (int kt = 0; kt < numKt - 1; ++kt) {
    int kn = kt + 1;
    bfrag a_n = *(const bfrag*)(aBase + (long)kn * 512);
    bfrag b_n0 = *(const bfrag*)(bBase + ((long)(nt0 + 0) * numKt + kn) * 512);
    bfrag b_n1 = *(const bfrag*)(bBase + ((long)(nt0 + 1) * numKt + kn) * 512);
    bfrag b_n2 = *(const bfrag*)(bBase + ((long)(nt0 + 2) * numKt + kn) * 512);
    bfrag b_n3 = *(const bfrag*)(bBase + ((long)(nt0 + 3) * numKt + kn) * 512);
    if (kn + 1 < numKt)  // speculative L2 prefetch two K-steps ahead
      __builtin_prefetch(aBase + (long)(kn + 1) * 512, 0, 1);
    acc[0] = wmma_bf16(a_c, b_c0, acc[0]);
    acc[1] = wmma_bf16(a_c, b_c1, acc[1]);
    acc[2] = wmma_bf16(a_c, b_c2, acc[2]);
    acc[3] = wmma_bf16(a_c, b_c3, acc[3]);
    a_c = a_n;
    b_c0 = b_n0; b_c1 = b_n1; b_c2 = b_n2; b_c3 = b_n3;
  }
  acc[0] = wmma_bf16(a_c, b_c0, acc[0]);
  acc[1] = wmma_bf16(a_c, b_c1, acc[1]);
  acc[2] = wmma_bf16(a_c, b_c2, acc[2]);
  acc[3] = wmma_bf16(a_c, b_c3, acc[3]);

#pragma unroll
  for (int j = 0; j < 4; ++j) {
    int col = (nt0 + j) * 16 + nlo;
    float bv = bias ? bias[col] : 0.f;
#pragma unroll
    for (int r = 0; r < 8; ++r) {
      int row = mt * 16 + r + half * 8;
      float v = acc[j][r] + bv;
      if (relu) v = fmaxf(v, 0.f);
      C[(long)row * ldc + col] = v;
    }
  }
}

// -------------------- layer-invariant blended attention --------------------
__device__ __forceinline__ float block_reduce(float v, float* red, int ismax) {
  int t = threadIdx.x;
  red[t] = v;
  __syncthreads();
  for (int s = 128; s > 0; s >>= 1) {
    if (t < s) {
      float o = red[t + s];
      red[t] = ismax ? fmaxf(red[t], o) : (red[t] + o);
    }
    __syncthreads();
  }
  float r = red[0];
  __syncthreads();
  return r;
}

// base[b,i,j] = (1-l3)(1-l1)l2*time_attn + (1-l3)l1*rel_attn + l3*resp_attn
__global__ __launch_bounds__(256) void base_attn_kernel(
    const float* __restrict__ rel, const float* __restrict__ resp,
    const float* __restrict__ ts, const float* lp1, const float* lp2,
    const float* lp3, float* __restrict__ baseo) {
  __shared__ float red[256];
  int i = blockIdx.x, b = blockIdx.y, t = threadIdx.x;
  long ro = ((long)b * L_ + i) * L_;
  float l1 = *lp1, l2 = *lp2, l3 = *lp3;
  float w_time = (1.f - l3) * (1.f - l1) * l2;
  float w_rel = (1.f - l3) * l1;
  float w_resp = l3;
  float xr[2], xs[2], xt[2];
#pragma unroll
  for (int u = 0; u < 2; ++u) {
    int j = t * 2 + u;
    bool fut = j > i;  // reference future mask (triu, k=1)
    float r = rel[ro + j];
    float rm = fut ? r : 0.f;
    xr[u] = (rm == 0.f) ? -1e4f : rm;
    float p = resp[ro + j];
    float pm = fut ? p : 0.f;
    xs[u] = (pm == 0.f) ? -1e4f : pm;
    float tv = __expf(-fabsf(ts[ro + j]));
    xt[u] = fut ? -__builtin_inff() : tv;
  }
  float mr = block_reduce(fmaxf(xr[0], xr[1]), red, 1);
  float er0 = __expf(xr[0] - mr), er1 = __expf(xr[1] - mr);
  float sr = block_reduce(er0 + er1, red, 0);
  float ms = block_reduce(fmaxf(xs[0], xs[1]), red, 1);
  float es0 = __expf(xs[0] - ms), es1 = __expf(xs[1] - ms);
  float ss = block_reduce(es0 + es1, red, 0);
  float mt = block_reduce(fmaxf(xt[0], xt[1]), red, 1);
  float et0 = __expf(xt[0] - mt), et1 = __expf(xt[1] - mt);
  float st = block_reduce(et0 + et1, red, 0);
  baseo[ro + t * 2] = w_rel * er0 / sr + w_resp * es0 / ss + w_time * et0 / st;
  baseo[ro + t * 2 + 1] =
      w_rel * er1 / sr + w_resp * es1 / ss + w_time * et1 / st;
}

// ----------------------- fused attention (per b,h) -------------------------
// block = 64 threads (2 waves), 32 query rows. S tile + packed-P tile in LDS.
__global__ __launch_bounds__(64) void attn_kernel(
    const __bf16* __restrict__ qAh,  // per (b,h): [mt(32)][kt(2)][lane][16]
    const __bf16* __restrict__ kBh,  // per (b,h): [nt(32)][kt(2)][lane][16]
    const __bf16* __restrict__ vBh,  // per (b,h): [nt(4)][kt(16)][lane][16]
    const float* __restrict__ base,  // (B,L,L)
    const float* lp1, const float* lp2, const float* lp3,
    float* __restrict__ outp,  // (B,L,E) f32
    int layer0) {
  __shared__ float S[32][512];                               // 64 KB
  __shared__ __attribute__((aligned(32))) __bf16 Pf[16384];  // 32 KB packed P
  __shared__ float red2[64];                                 // softmax combine
  int w = threadIdx.x >> 5, lane = threadIdx.x & 31;
  int h = blockIdx.y, b = blockIdx.z;
  int half = lane >> 4, nlo = lane & 15;
  long headOff = ((long)b * H_ + h) * ((long)L_ * D_);
  const __bf16* qA = qAh + headOff;
  const __bf16* kB = kBh + headOff;
  const __bf16* vB = vBh + headOff;
  int mt = blockIdx.x * 2 + w;

  // prefetch the blend-base rows this block will read in the softmax phase
  {
    const float* bbase = base + ((long)b * L_ + blockIdx.x * 32) * L_;
    for (int cl = threadIdx.x; cl < 32 * 512 / 32; cl += 64)
      __builtin_prefetch(bbase + (long)cl * 32, 0, 1);  // 128B granules
  }

  // ---- scores: S = (q k^T) / sqrt(d), causal "set to -1e9" mask ----
  bfrag aq0 = *(const bfrag*)(qA + ((long)(mt * 2 + 0) * 32 + lane) * 16);
  bfrag aq1 = *(const bfrag*)(qA + ((long)(mt * 2 + 1) * 32 + lane) * 16);
  bfrag kb0 = *(const bfrag*)(kB + ((long)(0 * 2 + 0) * 32 + lane) * 16);
  bfrag kb1 = *(const bfrag*)(kB + ((long)(0 * 2 + 1) * 32 + lane) * 16);
  for (int nt = 0; nt < 32; ++nt) {
    bfrag nb0 = kb0, nb1 = kb1;
    if (nt < 31) {  // double-buffer next K-tile pair
      nb0 = *(const bfrag*)(kB + ((long)((nt + 1) * 2 + 0) * 32 + lane) * 16);
      nb1 = *(const bfrag*)(kB + ((long)((nt + 1) * 2 + 1) * 32 + lane) * 16);
    }
    v8f acc = {};
    acc = wmma_bf16(aq0, kb0, acc);
    acc = wmma_bf16(aq1, kb1, acc);
    int col = nt * 16 + nlo;
#pragma unroll
    for (int r = 0; r < 8; ++r) {
      int rl = w * 16 + r + half * 8;   // local row 0..31
      int rowg = blockIdx.x * 32 + rl;  // query position 0..511
      float v = acc[r] * 0.125f;        // 1/sqrt(64)
      if (col > rowg) v = -1e9f;
      S[rl][col] = v;
    }
    kb0 = nb0;
    kb1 = nb1;
  }
  __syncthreads();

  // ---- softmax + blend with precomputed base, emit packed-A bf16 P ----
  // 2 threads per row: thread (rl, hf) owns columns [hf*256, hf*256+256)
  float l1 = *lp1, l2 = *lp2, l3 = *lp3;
  float c0 = (1.f - l3) * (1.f - l1) * (1.f - l2);
  {
    int rl = threadIdx.x >> 1, hf = threadIdx.x & 1;
    int j0 = hf * 256;
    float m = -1e30f;
    for (int j = j0; j < j0 + 256; ++j) m = fmaxf(m, S[rl][j]);
    red2[threadIdx.x] = m;
    __syncthreads();
    m = fmaxf(red2[rl * 2], red2[rl * 2 + 1]);
    __syncthreads();
    float s = 0.f;
    for (int j = j0; j < j0 + 256; ++j) {
      float e = __expf(S[rl][j] - m);
      S[rl][j] = e;
      s += e;
    }
    red2[threadIdx.x] = s;
    __syncthreads();
    s = red2[rl * 2] + red2[rl * 2 + 1];
    float inv = c0 / s;
    int rowg = blockIdx.x * 32 + rl;
    const float* brow = base + ((long)b * L_ + rowg) * L_;
    int wt = rl >> 4, mp = rl & 15;
    for (int j = j0; j < j0 + 256; ++j) {
      float p = S[rl][j] * inv + brow[j];
      int kt = j >> 5, kk = j & 31;
      int hi = (kk >> 4) & 1;
      int hfk = (kk >> 3) & 1;
      int slot = (hi ? 8 : 0) + (kk & 7);
      int ln = mp + hfk * 16;
      Pf[((wt * 16 + kt) * 32 + ln) * 16 + slot] = to_bf16(p);
    }
  }
  __syncthreads();

  // ---- out = P @ V (LDS A-fragments x global B-fragments) ----
  v8f o[4] = {{}, {}, {}, {}};
  bfrag vb0 = *(const bfrag*)(vB + ((long)(0 * 16 + 0) * 32 + lane) * 16);
  bfrag vb1 = *(const bfrag*)(vB + ((long)(1 * 16 + 0) * 32 + lane) * 16);
  bfrag vb2 = *(const bfrag*)(vB + ((long)(2 * 16 + 0) * 32 + lane) * 16);
  bfrag vb3 = *(const bfrag*)(vB + ((long)(3 * 16 + 0) * 32 + lane) * 16);
  for (int kt = 0; kt < 16; ++kt) {
    bfrag n0 = vb0, n1 = vb1, n2 = vb2, n3 = vb3;
    if (kt < 15) {  // double-buffer next V fragments
      int kn = kt + 1;
      n0 = *(const bfrag*)(vB + ((long)(0 * 16 + kn) * 32 + lane) * 16);
      n1 = *(const bfrag*)(vB + ((long)(1 * 16 + kn) * 32 + lane) * 16);
      n2 = *(const bfrag*)(vB + ((long)(2 * 16 + kn) * 32 + lane) * 16);
      n3 = *(const bfrag*)(vB + ((long)(3 * 16 + kn) * 32 + lane) * 16);
    }
    bfrag ap = *(const bfrag*)(&Pf[((w * 16 + kt) * 32 + lane) * 16]);
    o[0] = wmma_bf16(ap, vb0, o[0]);
    o[1] = wmma_bf16(ap, vb1, o[1]);
    o[2] = wmma_bf16(ap, vb2, o[2]);
    o[3] = wmma_bf16(ap, vb3, o[3]);
    vb0 = n0; vb1 = n1; vb2 = n2; vb3 = n3;
  }
#pragma unroll
  for (int nt = 0; nt < 4; ++nt)
#pragma unroll
    for (int r = 0; r < 8; ++r) {
      int rowg = blockIdx.x * 32 + w * 16 + r + half * 8;
      int colg = h * D_ + nt * 16 + nlo;
      float* dst = outp + ((long)b * L_ + rowg) * E_ + colg;
      float v = o[nt][r];
      if (layer0)
        *dst = v;  // layer 0: outputs = attn
      else
        *dst += fmaxf(v, 0.f);  // layer i: outputs += relu(residual)
    }
}

// ------------------------- final E->1 projection ---------------------------
__global__ __launch_bounds__(256) void final_dot_kernel(
    const float* __restrict__ hbuf, const float* __restrict__ w2,
    const float* __restrict__ b2, float* __restrict__ out) {
  int wv = threadIdx.x >> 5, lane = threadIdx.x & 31;
  int row = blockIdx.x * 8 + wv;
  const float* hr = hbuf + (long)row * E_;
  float s = 0.f;
  for (int j = lane; j < E_; j += 32) s += hr[j] * w2[j];
#pragma unroll
  for (int off = 16; off; off >>= 1) s += __shfl_xor(s, off, 32);
  if (lane == 0) out[row] = s + b2[0];
}

// ---------------------------------------------------------------------------
extern "C" void kernel_launch(void* const* d_in, const int* in_sizes, int n_in,
                              void* d_out, int out_size, void* d_ws,
                              size_t ws_size, hipStream_t stream) {
  (void)in_sizes; (void)n_in; (void)out_size; (void)ws_size;
  const float* inputs_4e   = (const float*)d_in[0];
  const float* query_embed = (const float*)d_in[1];
  const float* rel         = (const float*)d_in[2];
  const float* resp_rel    = (const float*)d_in[3];
  const float* timestamp   = (const float*)d_in[4];
  const float* l1          = (const float*)d_in[5];
  const float* l2          = (const float*)d_in[6];
  const float* l3          = (const float*)d_in[7];
  const float* Win         = (const float*)d_in[8];
  const float* b_in        = (const float*)d_in[9];
  const float* Wqkv        = (const float*)d_in[10];
  const float* bqkv        = (const float*)d_in[11];
  const float* Wf1         = (const float*)d_in[12];
  const float* bf1         = (const float*)d_in[13];
  const float* Wf2         = (const float*)d_in[14];
  const float* bf2         = (const float*)d_in[15];
  float* out = (float*)d_out;

  // workspace carve (all sizes are exact multiples of fragment tiles)
  char* wsp = (char*)d_ws;
  size_t off = 0;
  auto carve = [&](size_t bytes) -> void* {
    void* p = wsp + off;
    off += (bytes + 255) & ~(size_t)255;
    return p;
  };
  float* baseF    = (float*)carve((size_t)B_ * L_ * L_ * 4);
  float* inputsF  = (float*)carve((size_t)M_ * E_ * 4);
  float* outputsF = (float*)carve((size_t)M_ * E_ * 4);
  float* qF       = (float*)carve((size_t)M_ * E_ * 4);
  float* kF       = (float*)carve((size_t)M_ * E_ * 4);
  float* vF       = (float*)carve((size_t)M_ * E_ * 4);
  float* hF       = (float*)carve((size_t)M_ * E_ * 4);
  __bf16* WinB  = (__bf16*)carve((size_t)4 * E_ * E_ * 2);
  __bf16* WqkvB = (__bf16*)carve((size_t)6 * E_ * E_ * 2);
  __bf16* Wf1B  = (__bf16*)carve((size_t)2 * E_ * E_ * 2);
  __bf16* A4e   = (__bf16*)carve((size_t)M_ * 4 * E_ * 2);
  __bf16* qeA   = (__bf16*)carve((size_t)M_ * E_ * 2);
  __bf16* xA    = (__bf16*)carve((size_t)M_ * E_ * 2);
  __bf16* catA  = (__bf16*)carve((size_t)M_ * 2 * E_ * 2);
  __bf16* qAh   = (__bf16*)carve((size_t)B_ * H_ * L_ * D_ * 2);
  __bf16* kBh   = (__bf16*)carve((size_t)B_ * H_ * L_ * D_ * 2);
  __bf16* vBh   = (__bf16*)carve((size_t)B_ * H_ * L_ * D_ * 2);

  dim3 t256(256), t128(128), t64(64);
  long EE = (long)E_ * E_;

  // --- weight packing (B operands) ---
  pack_b_kernel<<<dim3(32 * 64 / 8, 1, 1), t256, 0, stream>>>(
      Win, WinB, E_, 32, 64, 0, 0, 0, 0, 0);
  pack_b_kernel<<<dim3(32 * 16 / 8, 6, 1), t256, 0, stream>>>(
      Wqkv, WqkvB, E_, 32, 16, 0, EE, EE, 0, 0);
  pack_b_kernel<<<dim3(32 * 32 / 8, 1, 1), t256, 0, stream>>>(
      Wf1, Wf1B, E_, 32, 32, 0, 0, 0, 0, 0);

  // --- activation packing (A operands) ---
  pack_a_kernel<<<dim3(256 * 64 / 8, 1, 1), t256, 0, stream>>>(
      inputs_4e, A4e, 4 * E_, 256, 64, 64, 0, 0, 0, 0, 0);
  pack_a_kernel<<<dim3(256 * 16 / 8, 1, 1), t256, 0, stream>>>(
      query_embed, qeA, E_, 256, 16, 16, 0, 0, 0, 0, 0);

  // inputs = relu(inputs_4e @ Win + b_in)
  gemm_wmma_kernel<<<dim3(E_ / 64, M_ / 64), t128, 0, stream>>>(
      A4e, WinB, b_in, inputsF, 64, E_, 1);

  // layer-invariant blended rel/resp/time attention base
  base_attn_kernel<<<dim3(L_, B_), t256, 0, stream>>>(
      rel, resp_rel, timestamp, l1, l2, l3, baseF);

  for (int layer = 0; layer < NL_; ++layer) {
    const float* kvsrc = (layer == 0) ? inputsF : outputsF;
    pack_a_kernel<<<dim3(256 * 16 / 8, 1, 1), t256, 0, stream>>>(
        kvsrc, xA, E_, 256, 16, 16, 0, 0, 0, 0, 0);
    const __bf16* Wq = WqkvB + (long)(layer * 3 + 0) * EE;
    const __bf16* Wk = WqkvB + (long)(layer * 3 + 1) * EE;
    const __bf16* Wv = WqkvB + (long)(layer * 3 + 2) * EE;
    const float* bq = bqkv + (long)(layer * 3 + 0) * E_;
    const float* bk = bqkv + (long)(layer * 3 + 1) * E_;
    const float* bv = bqkv + (long)(layer * 3 + 2) * E_;
    gemm_wmma_kernel<<<dim3(E_ / 64, M_ / 64), t128, 0, stream>>>(
        qeA, Wq, bq, qF, 16, E_, 0);
    gemm_wmma_kernel<<<dim3(E_ / 64, M_ / 64), t128, 0, stream>>>(
        xA, Wk, bk, kF, 16, E_, 0);
    gemm_wmma_kernel<<<dim3(E_ / 64, M_ / 64), t128, 0, stream>>>(
        xA, Wv, bv, vF, 16, E_, 0);
    // per-head repack: q -> A(512x64), k^T -> B(64x512), v -> B(512x64)
    long srcY = D_, srcZ = (long)L_ * E_;
    long dstY = (long)L_ * D_, dstZ = (long)H_ * L_ * D_;
    pack_a_kernel<<<dim3(8, H_, B_), t256, 0, stream>>>(
        qF, qAh, E_, 32, 2, 2, 0, srcY, dstY, srcZ, dstZ);
    pack_b_kernel<<<dim3(8, H_, B_), t256, 0, stream>>>(
        kF, kBh, E_, 32, 2, 1, srcY, dstY, srcZ, dstZ);
    pack_b_kernel<<<dim3(8, H_, B_), t256, 0, stream>>>(
        vF, vBh, E_, 4, 16, 0, srcY, dstY, srcZ, dstZ);
    attn_kernel<<<dim3(L_ / 32, H_, B_), t64, 0, stream>>>(
        qAh, kBh, vBh, baseF, l1, l2, l3, outputsF, layer == 0 ? 1 : 0);
  }

  // concat [outputs, query_embed] packed as one A operand (K = 1024)
  pack_a_kernel<<<dim3(256 * 16 / 8, 1, 1), t256, 0, stream>>>(
      outputsF, catA, E_, 256, 16, 32, 0, 0, 0, 0, 0);
  pack_a_kernel<<<dim3(256 * 16 / 8, 1, 1), t256, 0, stream>>>(
      query_embed, catA, E_, 256, 16, 32, 16, 0, 0, 0, 0);
  gemm_wmma_kernel<<<dim3(E_ / 64, M_ / 64), t128, 0, stream>>>(
      catA, Wf1B, bf1, hF, 32, E_, 1);
  final_dot_kernel<<<dim3(M_ / 8), t256, 0, stream>>>(hF, Wf2, bf2, out);
}
